// RNNmodel_19335942767149
// MI455X (gfx1250) — matrix operand order repaired and checked by reading, not compiled, not run on hardware
//
#include <hip/hip_runtime.h>

// ---------------------------------------------------------------------------
// 2-layer tanh RNN on MI455X (gfx1250, wave32, WMMA).
//   h0 = tanh(x2h0(x_t) + h2h0(h0));  h1 = tanh(x2h1(h0) + h2h1(h1))
// Strategy:
//   * hoist x2h0 over all T (parallel WMMA GEMM, bf16 in / f32 acc)
//   * persistent kernel for the 256 serial steps, grid barrier between GEMMs
//   * weights converted once to bf16 (L2 resident: 3x2MB)
//   * activation tiles staged to LDS via async global->LDS DMA (ASYNCcnt),
//     double-buffered so the H1 stage overlaps the layer-0 K-loop
// ---------------------------------------------------------------------------

typedef __attribute__((ext_vector_type(16))) __bf16 v16bf;
typedef __attribute__((ext_vector_type(8)))  __bf16 v8bf;
typedef __attribute__((ext_vector_type(8)))  float  v8f;
typedef __attribute__((ext_vector_type(4)))  int    v4i;

#define EMB    512
#define HID    1024
#define OUTD   4
#define BATCH  128
#define TSTEPS 256

#define NWG_RNN 32          // 8 row-blocks x 4 col-groups, 8 waves/WG, 2 tiles/wave
#define LDA_R   (HID + 8)   // LDS row stride (bf16) with 16B pad -> bank spread
#define LDA_P   (EMB + 8)

// -------- gfx1250 async global->LDS path (probe with fallback) -------------
#if defined(__AMDGCN__)
#  if __has_builtin(__builtin_amdgcn_global_load_async_to_lds_b128)
#    define ASYNC_LDS 1
#  endif
#endif

#ifdef ASYNC_LDS
typedef __attribute__((address_space(1))) v4i* as1_v4i;   // global (per diag)
typedef __attribute__((address_space(3))) v4i* as3_v4i;   // LDS
#  if __has_builtin(__builtin_amdgcn_s_wait_asynccnt)
#    define WAIT_ASYNC(N) __builtin_amdgcn_s_wait_asynccnt(N)
#  else
#    define WAIT_ASYNC(N) asm volatile("s_wait_asynccnt " #N ::: "memory")
#  endif
#else
#  define WAIT_ASYNC(N)
#endif

__device__ __forceinline__ float  bf2f(__bf16 b) { return (float)b; }
__device__ __forceinline__ __bf16 f2bf(float f)  { return (__bf16)f; }

// A fragment (16x32 bf16): lane m=lane&15 holds K = kb..kb+15,
// kb = (lane>=16)?16:0  (contiguous, matches B)
__device__ __forceinline__ v16bf make_a_frag(const __bf16* p) {
  v8bf lo = *(const v8bf*)p;
  v8bf hi = *(const v8bf*)(p + 8);
  return __builtin_shufflevector(lo, hi, 0,1,2,3,4,5,6,7,8,9,10,11,12,13,14,15);
}

__device__ __forceinline__ v8f wmma_bf16(v16bf a, v16bf b, v8f c) {
  return __builtin_amdgcn_wmma_f32_16x16x32_bf16(false, a, false, b, (short)0, c,
                                                 false, false);
}

// ---------------------------------------------------------------------------
__global__ __launch_bounds__(256) void cvt_bf16_kernel(const float* __restrict__ src,
                                                       unsigned short* __restrict__ dst,
                                                       int n) {
  int i = blockIdx.x * 256 + threadIdx.x;
  if (i < n) ((__bf16*)dst)[i] = f2bf(src[i]);
}

// ---------------------------------------------------------------------------
// preX[t*B+b][n] = emb[fx[b,t]] @ x2h_w0^T + x2h_b0 + h2h_b0   (stored bf16)
// grid: (B*T/16 row-blocks, HID/256 col-groups), 256 threads.
__global__ __launch_bounds__(256) void precompute_x0_kernel(
    const int* __restrict__ fx, const float* __restrict__ emb,
    const unsigned short* __restrict__ Wx0raw,   // [HID][EMB] bf16
    const float* __restrict__ bx0, const float* __restrict__ bh0,
    unsigned short* __restrict__ preXraw) {
  const __bf16* Wx0 = (const __bf16*)Wx0raw;
  __bf16* preX = (__bf16*)preXraw;

  __shared__ __bf16 As[16 * LDA_P];
  __shared__ int toks[16];

  const int g0 = blockIdx.x * 16;       // global row base (g = t*128 + b)
  const int t  = g0 >> 7;
  const int b0 = g0 & 127;
  const int lane = threadIdx.x & 31;
  const int w    = threadIdx.x >> 5;

  if (threadIdx.x < 16) toks[threadIdx.x] = fx[(size_t)(b0 + threadIdx.x) * TSTEPS + t];
  __syncthreads();
  // gather 16 embedding rows, f32 -> bf16, into LDS
  for (int i = threadIdx.x * 4; i < 16 * EMB; i += 256 * 4) {
    int r = i >> 9, c = i & (EMB - 1);
    float4 v = *(const float4*)(emb + (size_t)toks[r] * EMB + c);
    __bf16* d = &As[r * LDA_P + c];
    d[0] = f2bf(v.x); d[1] = f2bf(v.y); d[2] = f2bf(v.z); d[3] = f2bf(v.w);
  }
  __syncthreads();

  const int c0 = blockIdx.y * 16 + w * 2, c1 = c0 + 1;  // 16x16 col tiles
  const int m   = lane & 15;
  const int kbA = (lane >> 4) << 4;
  const int kbB = (lane >> 4) << 4;
  const int n   = lane & 15;
  v8f acc0 = {}, acc1 = {};
  for (int kt = 0; kt < EMB / 32; ++kt) {
    const int k0 = kt * 32;
    v16bf a  = make_a_frag(&As[m * LDA_P + k0 + kbA]);
    v16bf b0 = *(const v16bf*)(Wx0 + (size_t)(c0 * 16 + n) * EMB + k0 + kbB);
    v16bf b1 = *(const v16bf*)(Wx0 + (size_t)(c1 * 16 + n) * EMB + k0 + kbB);
    acc0 = wmma_bf16(a, b0, acc0);
    acc1 = wmma_bf16(a, b1, acc1);
  }
  const int moff = (lane >> 4) << 3;
  for (int r = 0; r < 8; ++r) {
    int gm = g0 + r + moff;
    int n0 = c0 * 16 + n, n1 = c1 * 16 + n;
    preX[(size_t)gm * HID + n0] = f2bf(acc0[r] + bx0[n0] + bh0[n0]);
    preX[(size_t)gm * HID + n1] = f2bf(acc1[r] + bx0[n1] + bh0[n1]);
  }
}

// ---------------------------------------------------------------------------
// Stage 16 rows x HID bf16 (32KB) global -> LDS.
// Async path: per-lane GLOBAL_LOAD_ASYNC_TO_LDS_B128, 8 issues per wave,
// tracked with ASYNCcnt (no VGPR round-trip). Fallback: load+ds_store.
__device__ __forceinline__ void stage_rows(const __bf16* __restrict__ g, __bf16* s) {
  for (int i = threadIdx.x; i < 16 * (HID / 8); i += 256) {
    int r = i >> 7;               // HID/8 = 128 chunks per row
    int c = (i & 127) << 3;
#ifdef ASYNC_LDS
    __builtin_amdgcn_global_load_async_to_lds_b128(
        (as1_v4i)(g + (size_t)r * HID + c),
        (as3_v4i)(s + r * LDA_R + c), 0, 0);
#else
    *(uint4*)(s + r * LDA_R + c) = *(const uint4*)(g + (size_t)r * HID + c);
#endif
  }
}

__device__ __forceinline__ void grid_sync(unsigned* bar) {
  __threadfence();
  __syncthreads();
  if (threadIdx.x == 0) {
    unsigned gen = __hip_atomic_load(&bar[1], __ATOMIC_RELAXED, __HIP_MEMORY_SCOPE_AGENT);
    unsigned arr = __hip_atomic_fetch_add(&bar[0], 1u, __ATOMIC_ACQ_REL,
                                          __HIP_MEMORY_SCOPE_AGENT);
    if (arr == NWG_RNN - 1) {
      __hip_atomic_store(&bar[0], 0u, __ATOMIC_RELAXED, __HIP_MEMORY_SCOPE_AGENT);
      __hip_atomic_fetch_add(&bar[1], 1u, __ATOMIC_RELEASE, __HIP_MEMORY_SCOPE_AGENT);
    } else {
      while (__hip_atomic_load(&bar[1], __ATOMIC_ACQUIRE, __HIP_MEMORY_SCOPE_AGENT) == gen)
        __builtin_amdgcn_s_sleep(2);
    }
  }
  __syncthreads();
}

// Persistent recurrent kernel: 32 WGs x 256 threads, 256 timesteps.
__global__ __launch_bounds__(256) void rnn_steps_kernel(
    const unsigned short* __restrict__ preXraw,
    const unsigned short* __restrict__ Wh0raw,  // h2h_w0 bf16 [HID][HID]
    const unsigned short* __restrict__ Wx1raw,  // x2h_w1
    const unsigned short* __restrict__ Wh1raw,  // h2h_w1
    const float* __restrict__ bx1, const float* __restrict__ bh1,
    unsigned short* __restrict__ H0raw,         // 2 ping-pong buffers
    unsigned short* __restrict__ H1raw,
    unsigned* __restrict__ bar) {
  const __bf16* preX = (const __bf16*)preXraw;
  const __bf16* Wh0  = (const __bf16*)Wh0raw;
  const __bf16* Wx1  = (const __bf16*)Wx1raw;
  const __bf16* Wh1  = (const __bf16*)Wh1raw;
  __bf16* H0 = (__bf16*)H0raw;
  __bf16* H1 = (__bf16*)H1raw;

  __shared__ __bf16 AsA[16 * LDA_R];   // layer-0 / layer-1 x-path activations
  __shared__ __bf16 AsB[16 * LDA_R];   // layer-1 h-path activations (prefetched)

  const int rb = blockIdx.x & 7;        // row block: batch rows rb*16..+15
  const int cg = blockIdx.x >> 3;       // col group: cols cg*256..+255
  const int w    = threadIdx.x >> 5;
  const int lane = threadIdx.x & 31;
  const int c0 = cg * 16 + w * 2, c1 = c0 + 1;
  const int m    = lane & 15;
  const int kbA  = (lane >> 4) << 4;
  const int kbB  = (lane >> 4) << 4;
  const int n    = lane & 15;
  const int moff = (lane >> 4) << 3;
  const size_t HSZ = (size_t)BATCH * HID;
  const v8f vzero = {};

  for (int t = 0; t < TSTEPS; ++t) {
    const int p = t & 1, q = p ^ 1;

    // Issue both stages up front: AsA <- H0[p] (needed now),
    // AsB <- H1[p] (needed in layer 1; overlaps the layer-0 K-loop).
    stage_rows(H0 + p * HSZ + (size_t)rb * 16 * HID, AsA);
    stage_rows(H1 + p * HSZ + (size_t)rb * 16 * HID, AsB);
    WAIT_ASYNC(8);                 // async loads retire in order: AsA resident
    __syncthreads();

    // ---------------- layer 0: H0[q] = tanh(H0[p] @ Wh0^T + preX[t]) --------
    v8f acc0 = vzero, acc1 = vzero;
    for (int kt = 0; kt < HID / 32; ++kt) {
      const int k0 = kt * 32;
      v16bf a = make_a_frag(&AsA[m * LDA_R + k0 + kbA]);
      const __bf16* pb0 = Wh0 + (size_t)(c0 * 16 + n) * HID + k0 + kbB;
      const __bf16* pb1 = Wh0 + (size_t)(c1 * 16 + n) * HID + k0 + kbB;
      v16bf b0 = *(const v16bf*)pb0;
      v16bf b1 = *(const v16bf*)pb1;
      __builtin_prefetch(pb0 + 32, 0, 3);   // global_prefetch next K chunk
      acc0 = wmma_bf16(a, b0, acc0);
      acc1 = wmma_bf16(a, b1, acc1);
    }
    for (int r = 0; r < 8; ++r) {
      int gm = rb * 16 + r + moff;
      int n0 = c0 * 16 + n, n1 = c1 * 16 + n;
      const size_t prow = ((size_t)t * BATCH + gm) * HID;
      H0[q * HSZ + (size_t)gm * HID + n0] = f2bf(tanhf(acc0[r] + bf2f(preX[prow + n0])));
      H0[q * HSZ + (size_t)gm * HID + n1] = f2bf(tanhf(acc1[r] + bf2f(preX[prow + n1])));
    }
    grid_sync(bar);

    // ------- layer 1: H1[q] = tanh(H0[q] @ Wx1^T + H1[p] @ Wh1^T + b) -------
    stage_rows(H0 + q * HSZ + (size_t)rb * 16 * HID, AsA);  // fresh layer-0 out
    WAIT_ASYNC(0);                 // AsA and AsB both resident
    __syncthreads();
    acc0 = vzero; acc1 = vzero;
    for (int kt = 0; kt < HID / 32; ++kt) {
      const int k0 = kt * 32;
      v16bf a  = make_a_frag(&AsA[m * LDA_R + k0 + kbA]);
      v16bf b0 = *(const v16bf*)(Wx1 + (size_t)(c0 * 16 + n) * HID + k0 + kbB);
      v16bf b1 = *(const v16bf*)(Wx1 + (size_t)(c1 * 16 + n) * HID + k0 + kbB);
      acc0 = wmma_bf16(a, b0, acc0);
      acc1 = wmma_bf16(a, b1, acc1);
    }
    // h-path: A already prefetched into AsB at top of step — no restage bubble
    for (int kt = 0; kt < HID / 32; ++kt) {
      const int k0 = kt * 32;
      v16bf a  = make_a_frag(&AsB[m * LDA_R + k0 + kbA]);
      v16bf b0 = *(const v16bf*)(Wh1 + (size_t)(c0 * 16 + n) * HID + k0 + kbB);
      v16bf b1 = *(const v16bf*)(Wh1 + (size_t)(c1 * 16 + n) * HID + k0 + kbB);
      acc0 = wmma_bf16(a, b0, acc0);
      acc1 = wmma_bf16(a, b1, acc1);
    }
    for (int r = 0; r < 8; ++r) {
      int gm = rb * 16 + r + moff;
      int n0 = c0 * 16 + n, n1 = c1 * 16 + n;
      H1[q * HSZ + (size_t)gm * HID + n0] = f2bf(tanhf(acc0[r] + bx1[n0] + bh1[n0]));
      H1[q * HSZ + (size_t)gm * HID + n1] = f2bf(tanhf(acc1[r] + bx1[n1] + bh1[n1]));
    }
    grid_sync(bar);
  }
}

// ---------------------------------------------------------------------------
__global__ __launch_bounds__(512) void fc_kernel(const unsigned short* __restrict__ H1raw,
                                                 const float* __restrict__ fw,
                                                 const float* __restrict__ fb,
                                                 float* __restrict__ out) {
  const __bf16* h1 = (const __bf16*)H1raw;
  int tid = threadIdx.x;
  if (tid >= BATCH * OUTD) return;
  int b = tid >> 2, o = tid & 3;
  float acc = fb[o];
  const __bf16* h = h1 + (size_t)b * HID;
  const float* wr = fw + (size_t)o * HID;
  for (int k = 0; k < HID; ++k) acc += bf2f(h[k]) * wr[k];
  out[tid] = acc;
}

// ---------------------------------------------------------------------------
extern "C" void kernel_launch(void* const* d_in, const int* in_sizes, int n_in,
                              void* d_out, int out_size, void* d_ws, size_t ws_size,
                              hipStream_t stream) {
  const int*   fx     = (const int*)  d_in[0];
  const float* emb    = (const float*)d_in[1];
  const float* x2h_w0 = (const float*)d_in[2];
  const float* x2h_b0 = (const float*)d_in[3];
  const float* h2h_w0 = (const float*)d_in[4];
  const float* h2h_b0 = (const float*)d_in[5];
  const float* x2h_w1 = (const float*)d_in[6];
  const float* x2h_b1 = (const float*)d_in[7];
  const float* h2h_w1 = (const float*)d_in[8];
  const float* h2h_b1 = (const float*)d_in[9];
  const float* fc_w   = (const float*)d_in[10];
  const float* fc_b   = (const float*)d_in[11];

  char* ws = (char*)d_ws;
  size_t off = 0;
  unsigned*       bar  = (unsigned*)(ws + off);        off += 256;
  unsigned short* Wh0  = (unsigned short*)(ws + off);  off += (size_t)HID * HID * 2;
  unsigned short* Wx1  = (unsigned short*)(ws + off);  off += (size_t)HID * HID * 2;
  unsigned short* Wh1  = (unsigned short*)(ws + off);  off += (size_t)HID * HID * 2;
  unsigned short* Wx0  = (unsigned short*)(ws + off);  off += (size_t)HID * EMB * 2;
  unsigned short* H0   = (unsigned short*)(ws + off);  off += (size_t)2 * BATCH * HID * 2;
  unsigned short* H1   = (unsigned short*)(ws + off);  off += (size_t)2 * BATCH * HID * 2;
  unsigned short* preX = (unsigned short*)(ws + off);  off += (size_t)TSTEPS * BATCH * HID * 2;

  // reset barrier + zero-init hidden state (bf16 zero == 0x0000)
  (void)hipMemsetAsync(bar, 0, 256, stream);
  (void)hipMemsetAsync(H0, 0, (size_t)2 * BATCH * HID * 2, stream);
  (void)hipMemsetAsync(H1, 0, (size_t)2 * BATCH * HID * 2, stream);

  // f32 -> bf16 weight conversion (L2-resident thereafter)
  cvt_bf16_kernel<<<(HID * HID + 255) / 256, 256, 0, stream>>>(h2h_w0, Wh0, HID * HID);
  cvt_bf16_kernel<<<(HID * HID + 255) / 256, 256, 0, stream>>>(x2h_w1, Wx1, HID * HID);
  cvt_bf16_kernel<<<(HID * HID + 255) / 256, 256, 0, stream>>>(h2h_w1, Wh1, HID * HID);
  cvt_bf16_kernel<<<(HID * EMB + 255) / 256, 256, 0, stream>>>(x2h_w0, Wx0, HID * EMB);

  // hoisted input projection for all 32768 (t,b) rows
  precompute_x0_kernel<<<dim3((BATCH * TSTEPS) / 16, HID / 256), 256, 0, stream>>>(
      fx, emb, Wx0, x2h_b0, h2h_b0, preX);

  // persistent serial recurrence (512 grid barriers)
  rnn_steps_kernel<<<NWG_RNN, 256, 0, stream>>>(preX, Wh0, Wx1, Wh1, x2h_b1, h2h_b1,
                                                H0, H1, bar);

  // final FC on last-step h1 (lands in ping-pong buffer 0 since T is even)
  fc_kernel<<<1, 512, 0, stream>>>(H1, fc_w, fc_b, (float*)d_out);
}
